// NEDMP_48962627174427
// MI455X (gfx1250) — matrix operand order repaired and checked by reading, not compiled, not run on hardware
//
#include <hip/hip_runtime.h>
#include <hip/hip_bf16.h>
#include <math.h>

typedef __attribute__((ext_vector_type(16))) _Float16 v16h;
typedef __attribute__((ext_vector_type(8)))  float    v8f;

#define HIDDEN 32

__device__ __forceinline__ float sigm(float x) { return 1.0f / (1.0f + __expf(-x)); }

// Pack A fragment (16x32 f16): lane l -> row l&15; element i (0..15):
//   i in [0,8)  -> K = g*8 + i
//   i in [8,16) -> K = 16 + g*8 + (i-8)
// xr must point at X_row + g*8 (16B aligned).
__device__ __forceinline__ v16h packA(const float* __restrict__ xr) {
    float4 x0 = *(const float4*)(xr);
    float4 x1 = *(const float4*)(xr + 4);
    float4 x2 = *(const float4*)(xr + 16);
    float4 x3 = *(const float4*)(xr + 20);
    v16h a;
    a[0]  = (_Float16)x0.x; a[1]  = (_Float16)x0.y; a[2]  = (_Float16)x0.z; a[3]  = (_Float16)x0.w;
    a[4]  = (_Float16)x1.x; a[5]  = (_Float16)x1.y; a[6]  = (_Float16)x1.z; a[7]  = (_Float16)x1.w;
    a[8]  = (_Float16)x2.x; a[9]  = (_Float16)x2.y; a[10] = (_Float16)x2.z; a[11] = (_Float16)x2.w;
    a[12] = (_Float16)x3.x; a[13] = (_Float16)x3.y; a[14] = (_Float16)x3.z; a[15] = (_Float16)x3.w;
    return a;
}

// Pack B fragment (32x16 f16): lane l -> col n; element i<8 -> K = g*8 + i, i>=8 -> K = 16 + g*8 + (i-8) (kbase = g*16)
template <int NC>
__device__ __forceinline__ v16h packB(const float* __restrict__ W, int kbase, int n) {
    v16h b;
    #pragma unroll
    for (int v = 0; v < 8; v++) {
        b[v]     = (_Float16)W[(size_t)(kbase / 2 + v) * NC + n];
        b[8 + v] = (_Float16)W[(size_t)(16 + kbase / 2 + v) * NC + n];
    }
    return b;
}

// ---------------- generic zero ----------------
__global__ void k_zero(float* __restrict__ p, int n) {
    int i = blockIdx.x * blockDim.x + threadIdx.x;
    if (i < n) p[i] = 0.0f;
}

// ---------------- init ----------------
__global__ void k_init_nodes(const float* __restrict__ gamma, const float* __restrict__ seeds,
                             float* __restrict__ Ps0, float* __restrict__ Pr,
                             float* __restrict__ logsum, float* __restrict__ deltas, int N) {
    int i = blockIdx.x * blockDim.x + threadIdx.x;
    if (i < N) {
        float s = seeds[i];
        Ps0[i] = 1.0f - s;
        Pr[i]  = gamma[i] * s;
        logsum[i] = 0.0f;
    }
    if (i < 6) deltas[i] = 0.0f;
}

__global__ void k_init_edges(const float* __restrict__ w, const float* __restrict__ gamma,
                             const int* __restrict__ src, const int* __restrict__ tar,
                             const float* __restrict__ Ps0,
                             float* __restrict__ theta, float* __restrict__ Ps_i0,
                             float* __restrict__ w1g, float* __restrict__ logsum, int E) {
    int e = blockIdx.x * blockDim.x + threadIdx.x;
    if (e >= E) return;
    int s = src[e];
    float p0   = Ps0[s];
    float phi0 = 1.0f - p0;
    float we   = w[e];
    float th   = 1.0f - we * phi0 + 1e-20f;
    theta[e] = th;
    Ps_i0[e] = p0;
    w1g[e]   = (1.0f - we) * (1.0f - gamma[s]);
    atomicAdd(&logsum[tar[e]], logf(th));
}

__global__ void k_init_edges2(const int* __restrict__ src, const int* __restrict__ cave,
                              const float* __restrict__ logsum,
                              const float* __restrict__ theta, const float* __restrict__ Ps_i0,
                              const float* __restrict__ w1g,
                              float* __restrict__ ps_ij, float* __restrict__ phi,
                              float* __restrict__ hidden,
                              const float* __restrict__ tw, const float* __restrict__ tb, int E) {
    int e = blockIdx.x * blockDim.x + threadIdx.x;
    if (e >= E) return;
    float p0   = Ps_i0[e];
    float phi0 = 1.0f - p0;
    float mm = __expf(logsum[src[e]]) / theta[cave[e]];
    float ps = p0 * mm;
    float ph = w1g[e] * phi0 - (ps - p0);
    ps_ij[e] = ps;
    phi[e]   = ph;
    float th = theta[e];
    float* h = hidden + (size_t)e * HIDDEN;
    #pragma unroll
    for (int j = 0; j < HIDDEN; j++) {
        float v = th * tw[j] + ph * tw[HIDDEN + j] + ps * tw[2 * HIDDEN + j] + tb[j];
        h[j] = fmaxf(v, 0.0f);
    }
}

__global__ void k_init_marg(const float* __restrict__ Ps0, const float* __restrict__ logsum,
                            const float* __restrict__ Pr_in, float* __restrict__ Ps,
                            float* __restrict__ out, int N) {
    int n = blockIdx.x * blockDim.x + threadIdx.x;
    if (n >= N) return;
    float ps = Ps0[n] * __expf(logsum[n]);
    Ps[n] = ps;
    float pr = Pr_in[n];
    float pi = 1.0f - ps - pr;
    out[n * 3 + 0] = ps;
    out[n * 3 + 1] = pi;
    out[n * 3 + 2] = pr;
}

// ---------------- per-layer elementwise ----------------
__global__ void k_layer_theta(const float* __restrict__ w, const int* __restrict__ tar,
                              float* __restrict__ theta, const float* __restrict__ phi,
                              float* __restrict__ logsum, int E) {
    int e = blockIdx.x * blockDim.x + threadIdx.x;
    if (e >= E) return;
    float th = theta[e] - w[e] * phi[e];
    theta[e] = th;
    atomicAdd(&logsum[tar[e]], logf(th));
}

__global__ void k_edge_msg(const int* __restrict__ src, const int* __restrict__ cave,
                           const float* __restrict__ logsum, const float* __restrict__ theta,
                           float* __restrict__ edge_msg, int E) {
    int e = blockIdx.x * blockDim.x + threadIdx.x;
    if (e >= E) return;
    edge_msg[e] = __expf(logsum[src[e]]) / theta[cave[e]];
}

__global__ void k_node_update(const float* __restrict__ logsum, const float* __restrict__ node_agg,
                              const float* __restrict__ m1w, const float* __restrict__ m1b,
                              const float* __restrict__ sd1w, const float* __restrict__ sd1b,
                              const float* __restrict__ Ps0, const float* __restrict__ gamma,
                              float* __restrict__ Ps, float* __restrict__ Pr,
                              float* __restrict__ node_msg, float* __restrict__ out,
                              float* __restrict__ deltas, int N, int layer) {
    int n = blockIdx.x * blockDim.x + threadIdx.x;
    if (n >= N) return;
    float raw = __expf(logsum[n]);
    float a0 = sd1b[0], a1 = sd1b[1];
    const float* na = node_agg + (size_t)n * HIDDEN;
    #pragma unroll
    for (int j = 0; j < HIDDEN; j++) {
        float x = na[j];
        a0 += x * sd1w[j * 2 + 0];
        a1 += x * sd1w[j * 2 + 1];
        float nm = fmaxf(raw * m1w[j] + m1b[j], 0.0f);
        a0 += nm * sd1w[(HIDDEN + j) * 2 + 0];
        a1 += nm * sd1w[(HIDDEN + j) * 2 + 1];
    }
    float r0 = sigm(a0), r1 = sigm(a1);
    float nmsg = fminf(raw * r0 + r1, 1.0f);
    node_msg[n] = nmsg;
    float ps = Ps[n], pr = Pr[n];
    float pi = 1.0f - ps - pr;
    float ps_n = fminf(ps, Ps0[n] * nmsg);
    float pr_n = fmaxf(pr, pr + gamma[n] * pi);
    float pi_n = 1.0f - ps_n - pr_n;
    Ps[n] = ps_n;
    Pr[n] = pr_n;
    float* o        = out + (size_t)(layer + 1) * N * 3 + (size_t)n * 3;
    const float* op = out + (size_t)layer * N * 3 + (size_t)n * 3;
    float d = fmaxf(fmaxf(fabsf(ps_n - op[0]), fabsf(pi_n - op[1])), fabsf(pr_n - op[2]));
    o[0] = ps_n; o[1] = pi_n; o[2] = pr_n;
    atomicMax((unsigned int*)&deltas[layer], __float_as_uint(d));
}

// eres + edge message / ps_ij / phi update (GRU part in fused WMMA kernel)
__global__ void k_edge_res(const float* __restrict__ hidden_agg,
                           float* __restrict__ edge_msg, float* __restrict__ ps_ij,
                           float* __restrict__ phi,
                           const float* __restrict__ Ps_i0, const float* __restrict__ w1g,
                           const float* __restrict__ m2w, const float* __restrict__ m2b,
                           const float* __restrict__ sd2w, const float* __restrict__ sd2b, int E) {
    int e = blockIdx.x * blockDim.x + threadIdx.x;
    if (e >= E) return;
    const float* ha = hidden_agg + (size_t)e * HIDDEN;
    float emsg = edge_msg[e];
    float a0 = sd2b[0], a1 = sd2b[1];
    #pragma unroll
    for (int j = 0; j < HIDDEN; j++) {
        float x = ha[j];
        a0 += x * sd2w[j * 2 + 0];
        a1 += x * sd2w[j * 2 + 1];
        float em = fmaxf(emsg * m2w[j] + m2b[j], 0.0f);
        a0 += em * sd2w[(HIDDEN + j) * 2 + 0];
        a1 += em * sd2w[(HIDDEN + j) * 2 + 1];
    }
    float r0 = sigm(a0), r1 = sigm(a1);
    float emn = fminf(emsg * r0 + r1, 1.0f);
    edge_msg[e] = emn;
    float psp = ps_ij[e];
    float psn = Ps_i0[e] * emn;
    ps_ij[e]  = psn;
    phi[e]    = w1g[e] * phi[e] - (psn - psp);
}

__global__ void k_finalize(float* __restrict__ out, int total) {
    int i = blockIdx.x * blockDim.x + threadIdx.x;
    if (i >= total) return;
    float v = out[i];
    v = (v <= 0.0f) ? 1e-20f : (v > 1.0f ? 1.0f : v);
    out[i] = logf(v);
}

// ---------------- plain WMMA GEMM: Y = act(X @ W + b), X rows x 32 ----------------
template <int NT, int ACT>
__global__ void k_gemm_wmma(const float* __restrict__ X, const float* __restrict__ W,
                            const float* __restrict__ bias, float* __restrict__ Y, int rows) {
    const int NC = NT * 16;
    int lane = threadIdx.x & 31;
    int wave = threadIdx.x >> 5;
    int ncol = lane & 15;
    int g    = lane >> 4;
    v16h B[NT];
    float bv[NT];
    #pragma unroll
    for (int nt = 0; nt < NT; nt++) {
        B[nt]  = packB<NC>(W, g * 16, nt * 16 + ncol);
        bv[nt] = bias[nt * 16 + ncol];
    }
    int nTiles     = rows >> 4;
    int wavesTotal = gridDim.x * (blockDim.x >> 5);
    for (int t = blockIdx.x * (blockDim.x >> 5) + wave; t < nTiles; t += wavesTotal) {
        int mrow = t * 16 + ncol;
        v16h A = packA(X + (size_t)mrow * 32 + g * 8);
        #pragma unroll
        for (int nt = 0; nt < NT; nt++) {
            v8f c = {};
            c = __builtin_amdgcn_wmma_f32_16x16x32_f16(false, A, false, B[nt], (short)0, c, false, false);
            int n = nt * 16 + ncol;
            #pragma unroll
            for (int v = 0; v < 8; v++) {
                int m = t * 16 + v + (g ? 8 : 0);
                float val = c[v] + bv[nt];
                if (ACT == 1) val = fmaxf(val, 0.0f);
                Y[(size_t)m * NC + n] = val;
            }
        }
    }
}

// ---------------- fused h_cat GEMM with LDS-staged B fragments ----------------
// h_cat = relu([hidden | te(theta,phi,ps)] @ cat_w + b), te on the fly,
// fused segment-sum scatter into agg[tar].
__global__ void k_gemm_hcat(const float* __restrict__ hidden,
                            const float* __restrict__ theta, const float* __restrict__ phi,
                            const float* __restrict__ ps_ij,
                            const float* __restrict__ tw, const float* __restrict__ tb,
                            const float* __restrict__ W, const float* __restrict__ bias,
                            const int* __restrict__ tar,
                            float* __restrict__ Y, float* __restrict__ agg, int rows) {
    __shared__ v16h ldsB[4][32];   // [0,1]: B0 (x hidden), [2,3]: B1 (x te)
    int lane = threadIdx.x & 31;
    int wave = threadIdx.x >> 5;
    int ncol = lane & 15;
    int g    = lane >> 4;
    int nWaves = blockDim.x >> 5;
    for (int f = wave; f < 4; f += nWaves) {
        int nt = f & 1;
        const float* Wp = (f < 2) ? W : (W + 32 * 32);
        ldsB[f][lane] = packB<32>(Wp, g * 16, nt * 16 + ncol);
    }
    __syncthreads();
    float bv[2] = {bias[ncol], bias[16 + ncol]};
    // theta-MLP column weights for this lane's A-fragment K-set
    float tw0[16], tw1[16], tw2[16], tbv[16];
    #pragma unroll
    for (int i = 0; i < 8; i++) {
        int k  = g * 8 + i;
        int k2 = 16 + g * 8 + i;
        tw0[i] = tw[k];      tw0[8 + i] = tw[k2];
        tw1[i] = tw[32 + k]; tw1[8 + i] = tw[32 + k2];
        tw2[i] = tw[64 + k]; tw2[8 + i] = tw[64 + k2];
        tbv[i] = tb[k];      tbv[8 + i] = tb[k2];
    }
    int nTiles     = rows >> 4;
    int wavesTotal = gridDim.x * nWaves;
    for (int t = blockIdx.x * nWaves + wave; t < nTiles; t += wavesTotal) {
        asm volatile("" ::: "memory");   // keep LDS B-frag reads inside the loop
        int mrow = t * 16 + ncol;
        v16h Ah = packA(hidden + (size_t)mrow * 32 + g * 8);
        float th = theta[mrow], ph = phi[mrow], ps = ps_ij[mrow];
        v16h At;
        #pragma unroll
        for (int i = 0; i < 16; i++) {
            At[i] = (_Float16)fmaxf(th * tw0[i] + ph * tw1[i] + ps * tw2[i] + tbv[i], 0.0f);
        }
        int trow[8];
        #pragma unroll
        for (int v = 0; v < 8; v++) trow[v] = tar[t * 16 + v + (g ? 8 : 0)];
        #pragma unroll
        for (int nt = 0; nt < 2; nt++) {
            v16h b0 = ldsB[nt][lane];
            v16h b1 = ldsB[2 + nt][lane];
            v8f c = {};
            c = __builtin_amdgcn_wmma_f32_16x16x32_f16(false, Ah, false, b0, (short)0, c, false, false);
            c = __builtin_amdgcn_wmma_f32_16x16x32_f16(false, At, false, b1, (short)0, c, false, false);
            int n = nt * 16 + ncol;
            #pragma unroll
            for (int v = 0; v < 8; v++) {
                int m = t * 16 + v + (g ? 8 : 0);
                float val = fmaxf(c[v] + bv[nt], 0.0f);
                Y[(size_t)m * 32 + n] = val;
                atomicAdd(&agg[(size_t)trow[v] * 32 + n], val);
            }
        }
    }
}

// ---------------- fused hidden_agg GEMM: relu((agg[src] - h_cat[cave]) @ a2_w + b) ----------------
__global__ void k_gemm_aggdiff(const float* __restrict__ agg, const float* __restrict__ h_cat,
                               const int* __restrict__ src, const int* __restrict__ cave,
                               const float* __restrict__ W, const float* __restrict__ bias,
                               float* __restrict__ Y, int rows) {
    int lane = threadIdx.x & 31;
    int wave = threadIdx.x >> 5;
    int ncol = lane & 15;
    int g    = lane >> 4;
    v16h B[2];
    float bv[2];
    #pragma unroll
    for (int nt = 0; nt < 2; nt++) {
        B[nt]  = packB<32>(W, g * 16, nt * 16 + ncol);
        bv[nt] = bias[nt * 16 + ncol];
    }
    int nTiles     = rows >> 4;
    int wavesTotal = gridDim.x * (blockDim.x >> 5);
    for (int t = blockIdx.x * (blockDim.x >> 5) + wave; t < nTiles; t += wavesTotal) {
        int mrow = t * 16 + ncol;
        const float* pa = agg   + (size_t)src[mrow]  * 32 + g * 8;
        const float* pc = h_cat + (size_t)cave[mrow] * 32 + g * 8;
        float4 a0 = *(const float4*)(pa);      float4 c0 = *(const float4*)(pc);
        float4 a1 = *(const float4*)(pa + 4);  float4 c1 = *(const float4*)(pc + 4);
        float4 a2 = *(const float4*)(pa + 16); float4 c2 = *(const float4*)(pc + 16);
        float4 a3 = *(const float4*)(pa + 20); float4 c3 = *(const float4*)(pc + 20);
        v16h A;
        A[0]  = (_Float16)(a0.x - c0.x); A[1]  = (_Float16)(a0.y - c0.y);
        A[2]  = (_Float16)(a0.z - c0.z); A[3]  = (_Float16)(a0.w - c0.w);
        A[4]  = (_Float16)(a1.x - c1.x); A[5]  = (_Float16)(a1.y - c1.y);
        A[6]  = (_Float16)(a1.z - c1.z); A[7]  = (_Float16)(a1.w - c1.w);
        A[8]  = (_Float16)(a2.x - c2.x); A[9]  = (_Float16)(a2.y - c2.y);
        A[10] = (_Float16)(a2.z - c2.z); A[11] = (_Float16)(a2.w - c2.w);
        A[12] = (_Float16)(a3.x - c3.x); A[13] = (_Float16)(a3.y - c3.y);
        A[14] = (_Float16)(a3.z - c3.z); A[15] = (_Float16)(a3.w - c3.w);
        #pragma unroll
        for (int nt = 0; nt < 2; nt++) {
            v8f c = {};
            c = __builtin_amdgcn_wmma_f32_16x16x32_f16(false, A, false, B[nt], (short)0, c, false, false);
            int n = nt * 16 + ncol;
            #pragma unroll
            for (int v = 0; v < 8; v++) {
                int m = t * 16 + v + (g ? 8 : 0);
                Y[(size_t)m * 32 + n] = fmaxf(c[v] + bv[nt], 0.0f);
            }
        }
    }
}

// ---------------- fused GRU: 12 WMMAs + gating, hidden in place, LDS-staged weights ----------------
__global__ void k_gru_fused(const float* __restrict__ Xa,   // hidden_agg [rows][32]
                            float* __restrict__ H,          // hidden [rows][32] (read + write)
                            const float* __restrict__ Wi, const float* __restrict__ Wh,  // [32][96]
                            const float* __restrict__ bi, const float* __restrict__ bh,
                            int rows) {
    __shared__ v16h ldsB[12][32];   // f<6: Bi[f]; f>=6: Bh[f-6]
    int lane = threadIdx.x & 31;
    int wave = threadIdx.x >> 5;
    int ncol = lane & 15;
    int g    = lane >> 4;
    int nWaves = blockDim.x >> 5;
    for (int f = wave; f < 12; f += nWaves) {
        int nt = f % 6;
        const float* Wp = (f < 6) ? Wi : Wh;
        ldsB[f][lane] = packB<96>(Wp, g * 16, nt * 16 + ncol);
    }
    __syncthreads();
    float bvi[6], bvh[6];
    #pragma unroll
    for (int nt = 0; nt < 6; nt++) {
        bvi[nt] = bi[nt * 16 + ncol];
        bvh[nt] = bh[nt * 16 + ncol];
    }
    int nTiles     = rows >> 4;
    int wavesTotal = gridDim.x * nWaves;
    for (int t = blockIdx.x * nWaves + wave; t < nTiles; t += wavesTotal) {
        asm volatile("" ::: "memory");   // keep LDS B-frag reads inside the loop
        int mrow = t * 16 + ncol;
        v16h Aa = packA(Xa + (size_t)mrow * 32 + g * 8);
        v16h Ah = packA(H  + (size_t)mrow * 32 + g * 8);
        float rs[2][8];
        #pragma unroll
        for (int c = 0; c < 2; c++) {      // r gate: column tiles 0,1
            v16h wi = ldsB[c][lane];
            v16h wh = ldsB[6 + c][lane];
            v8f ci = {}, ch = {};
            ci = __builtin_amdgcn_wmma_f32_16x16x32_f16(false, Aa, false, wi, (short)0, ci, false, false);
            ch = __builtin_amdgcn_wmma_f32_16x16x32_f16(false, Ah, false, wh, (short)0, ch, false, false);
            #pragma unroll
            for (int v = 0; v < 8; v++) rs[c][v] = sigm(ci[v] + bvi[c] + ch[v] + bvh[c]);
        }
        float ns[2][8];
        #pragma unroll
        for (int c = 0; c < 2; c++) {      // n gate: column tiles 4,5 (consumes r)
            v16h wi = ldsB[4 + c][lane];
            v16h wh = ldsB[10 + c][lane];
            v8f ci = {}, ch = {};
            ci = __builtin_amdgcn_wmma_f32_16x16x32_f16(false, Aa, false, wi, (short)0, ci, false, false);
            ch = __builtin_amdgcn_wmma_f32_16x16x32_f16(false, Ah, false, wh, (short)0, ch, false, false);
            #pragma unroll
            for (int v = 0; v < 8; v++)
                ns[c][v] = tanhf(ci[v] + bvi[4 + c] + rs[c][v] * (ch[v] + bvh[4 + c]));
        }
        #pragma unroll
        for (int c = 0; c < 2; c++) {      // z gate + combine: column tiles 2,3
            v16h wi = ldsB[2 + c][lane];
            v16h wh = ldsB[8 + c][lane];
            v8f ci = {}, ch = {};
            ci = __builtin_amdgcn_wmma_f32_16x16x32_f16(false, Aa, false, wi, (short)0, ci, false, false);
            ch = __builtin_amdgcn_wmma_f32_16x16x32_f16(false, Ah, false, wh, (short)0, ch, false, false);
            #pragma unroll
            for (int v = 0; v < 8; v++) {
                float z = sigm(ci[v] + bvi[2 + c] + ch[v] + bvh[2 + c]);
                int m = t * 16 + v + (g ? 8 : 0);
                size_t idx = (size_t)m * 32 + c * 16 + ncol;
                float hold = H[idx];
                H[idx] = (1.0f - z) * ns[c][v] + z * hold;
            }
        }
    }
}

// ---------------- host launch ----------------
extern "C" void kernel_launch(void* const* d_in, const int* in_sizes, int n_in,
                              void* d_out, int out_size, void* d_ws, size_t ws_size,
                              hipStream_t stream) {
    const float* theta_w = (const float*)d_in[0];
    const float* theta_b = (const float*)d_in[1];
    const float* m1_w    = (const float*)d_in[2];
    const float* m1_b    = (const float*)d_in[3];
    const float* m2_w    = (const float*)d_in[4];
    const float* m2_b    = (const float*)d_in[5];
    const float* a1_w    = (const float*)d_in[6];
    const float* a1_b    = (const float*)d_in[7];
    const float* a2_w    = (const float*)d_in[8];
    const float* a2_b    = (const float*)d_in[9];
    const float* cat_w   = (const float*)d_in[10];
    const float* cat_b   = (const float*)d_in[11];
    const float* gru_wi  = (const float*)d_in[12];
    const float* gru_wh  = (const float*)d_in[13];
    const float* gru_bi  = (const float*)d_in[14];
    const float* gru_bh  = (const float*)d_in[15];
    const float* sd1_w   = (const float*)d_in[16];
    const float* sd1_b   = (const float*)d_in[17];
    const float* sd2_w   = (const float*)d_in[18];
    const float* sd2_b   = (const float*)d_in[19];
    const float* weights = (const float*)d_in[20];
    const float* gamma   = (const float*)d_in[21];
    const float* seeds   = (const float*)d_in[22];
    const int*   src     = (const int*)d_in[23];
    const int*   tar     = (const int*)d_in[24];
    const int*   cave    = (const int*)d_in[25];

    const int E = in_sizes[20];
    const int N = in_sizes[21];
    const int L = 6;  // number_layers (fixed by the reference; device scalar unreadable under capture)

    float* ws = (float*)d_ws;
    size_t eE = (size_t)E;
    float* theta    = ws;
    float* phi      = ws + 1 * eE;
    float* ps_ij    = ws + 2 * eE;
    float* edge_msg = ws + 3 * eE;
    float* Ps_i0    = ws + 4 * eE;
    float* w1g      = ws + 5 * eE;
    float* hidden   = ws + 6 * eE;            // 32E
    float* h_cat    = ws + 38 * eE;           // 32E
    float* bufA     = ws + 70 * eE;           // hidden_agg (32E)
    float* nodebase = ws + 102 * eE;
    float* logsum   = nodebase;
    float* node_msg = nodebase + 1 * (size_t)N;
    float* Ps       = nodebase + 2 * (size_t)N;
    float* Pr       = nodebase + 3 * (size_t)N;
    float* Ps0      = nodebase + 4 * (size_t)N;
    float* agg_n    = nodebase + 5 * (size_t)N;   // 32N
    float* node_agg = nodebase + 37 * (size_t)N;  // 32N

    float* out    = (float*)d_out;
    float* deltas = out + (size_t)(L + 1) * N * 3;

    const int TPB = 256;
    int gE = (E + TPB - 1) / TPB;
    int gN = (N + TPB - 1) / TPB;
    dim3 gemmBlk(128);
    int gemmGridE = 2048;
    int gemmGridN = 640;

    // ---- init ----
    k_init_nodes<<<gN, TPB, 0, stream>>>(gamma, seeds, Ps0, Pr, logsum, deltas, N);
    k_init_edges<<<gE, TPB, 0, stream>>>(weights, gamma, src, tar, Ps0, theta, Ps_i0, w1g, logsum, E);
    k_init_edges2<<<gE, TPB, 0, stream>>>(src, cave, logsum, theta, Ps_i0, w1g, ps_ij, phi, hidden,
                                          theta_w, theta_b, E);
    k_init_marg<<<gN, TPB, 0, stream>>>(Ps0, logsum, Pr, Ps, out, N);

    // ---- layers ----
    for (int l = 0; l < L; l++) {
        k_zero<<<gN, TPB, 0, stream>>>(logsum, N);
        k_zero<<<(N * 32 + TPB - 1) / TPB, TPB, 0, stream>>>(agg_n, N * 32);

        k_layer_theta<<<gE, TPB, 0, stream>>>(weights, tar, theta, phi, logsum, E);
        k_edge_msg<<<gE, TPB, 0, stream>>>(src, cave, logsum, theta, edge_msg, E);

        // h_cat = relu([hidden | te] @ cat_w + cat_b), te on the fly, fused scatter
        k_gemm_hcat<<<gemmGridE, gemmBlk, 0, stream>>>(hidden, theta, phi, ps_ij,
                                                       theta_w, theta_b, cat_w, cat_b,
                                                       tar, h_cat, agg_n, E);

        // node_agg = relu(agg_n @ a1_w + a1_b)
        k_gemm_wmma<2, 1><<<gemmGridN, gemmBlk, 0, stream>>>(agg_n, a1_w, a1_b, node_agg, N);

        k_node_update<<<gN, TPB, 0, stream>>>(logsum, node_agg, m1_w, m1_b, sd1_w, sd1_b,
                                              Ps0, gamma, Ps, Pr, node_msg, out, deltas, N, l);

        // hidden_agg = relu((agg_n[src] - h_cat[cave]) @ a2_w + a2_b), gather fused
        k_gemm_aggdiff<<<gemmGridE, gemmBlk, 0, stream>>>(agg_n, h_cat, src, cave,
                                                          a2_w, a2_b, bufA, E);

        // GRU fully fused: hidden = gru(hidden_agg, hidden)
        k_gru_fused<<<gemmGridE, gemmBlk, 0, stream>>>(bufA, hidden, gru_wi, gru_wh,
                                                       gru_bi, gru_bh, E);

        // eres + edge message / ps_ij / phi updates
        k_edge_res<<<gE, TPB, 0, stream>>>(bufA, edge_msg, ps_ij, phi,
                                           Ps_i0, w1g, m2_w, m2_b, sd2_w, sd2_b, E);
    }

    // ---- clamp + log marginals (deltas slots untouched) ----
    int totalMarg = (L + 1) * N * 3;
    k_finalize<<<(totalMarg + TPB - 1) / TPB, TPB, 0, stream>>>(out, totalMarg);
}
